// VQVAELayer_82789789597696
// MI455X (gfx1250) — matrix operand order repaired and checked by reading, not compiled, not run on hardware
//
#include <hip/hip_runtime.h>

// ---------------------------------------------------------------------------
// VQ-VAE nearest-codebook lookup for MI455X (gfx1250, wave32, WMMA).
// argmin_j ||x - w_j||^2  ==  argmax_j ( x . w_j - 0.5*||w_j||^2 )
// GEMM part runs on v_wmma_f32_16x16x32_f16 (f16 inputs, f32 accumulate);
// the per-column bias 0.5*||w_j||^2 is precomputed in exact fp32.
// Codebook is pre-packed into WMMA B-fragments (contiguous k order) and held in
// LDS (132 KB of the 320 KB/WGP), amortized over 512 tokens per workgroup.
// ---------------------------------------------------------------------------

typedef __attribute__((ext_vector_type(16))) _Float16 v16h;
typedef __attribute__((ext_vector_type(8)))  float    v8f;
typedef __attribute__((ext_vector_type(4)))  float    f32x4;

#define EMBED       64
#define NCODE       1024
#define NTOK        262144
#define NTILES      (NCODE / 16)          // 64 code tiles of 16 columns
#define GBLK        4                     // 16-token blocks per wave
#define WAVES       8
#define TPB         (WAVES * 32)          // 256 threads
#define TOK_PER_WG  (WAVES * GBLK * 16)   // 512 tokens per workgroup

#define WP_HALFS    (NTILES * 2 * 32 * 16)   // 65536 f16 = 128 KiB packed B
#define WP_BYTES    (WP_HALFS * 2)           // 131072
#define WSQ_BYTES   (NCODE * 4)              // 4096  (0.5*||w_j||^2, fp32)
#define SMEM_BYTES  (WP_BYTES + WSQ_BYTES)   // 135168

// ---------------------------------------------------------------------------
// Prep kernel: pack w (fp32, [64 x 1024] row-major, w[d*1024+k]) into f16
// B-fragments, and compute 0.5*||w_j||^2 in fp32.
//
// B-fragment layout used here for V_WMMA_F32_16X16X32_F16 (B is 32x16, KxN), in
// contiguous k order, not the ISA table order (A uses the same; K halves split
// lanes 0-15 / 16-31):  half h of lane l holds  B[K][n] with
//     n = (l & 15),   K = h + (l >= 16 ? 16 : 0)
// Fragment storage:  wp[ ((tile*2 + kk)*32 + lane)*16 + h ],  kk = K-base/32.
// ---------------------------------------------------------------------------
__global__ void vq_pack_kernel(const float* __restrict__ w,
                               _Float16* __restrict__ wp,
                               float* __restrict__ wsq) {
  const int tid = blockIdx.x * blockDim.x + threadIdx.x;
  if (tid < WP_HALFS) {
    const int h  = tid & 15;
    const int l  = (tid >> 4) & 31;
    const int kk = (tid >> 9) & 1;
    const int t  = tid >> 10;
    const int K  = kk * 32 + ((l >= 16) ? 16 : 0) + h;
    const int n  = t * 16 + (l & 15);
    wp[tid] = (_Float16)w[K * NCODE + n];
  }
  if (tid < NCODE) {
    float s = 0.f;
#pragma unroll 8
    for (int d = 0; d < EMBED; ++d) {
      const float v = w[d * NCODE + tid];
      s += v * v;
    }
    wsq[tid] = 0.5f * s;
  }
}

// ---------------------------------------------------------------------------
// Main kernel.
// ---------------------------------------------------------------------------
__global__ __launch_bounds__(TPB) void vq_main_kernel(
    const float* __restrict__ x, const float* __restrict__ w,
    const _Float16* __restrict__ wp, float* __restrict__ out) {
  extern __shared__ char smem[];
  _Float16* lb  = (_Float16*)smem;             // packed B fragments (128 KiB)
  float*    lsq = (float*)(smem + WP_BYTES);   // 0.5*||w_j||^2     (4 KiB)

  const int tid   = threadIdx.x;
  const int lane  = tid & 31;
  const int wave  = tid >> 5;
  const int lan16 = lane & 15;
  const int koff  = (lane >> 4) * 16;          // A-fragment K offset per half

  // Cooperative fill of LDS: packed codebook + biases are contiguous in ws.
  {
    const uint4* src = (const uint4*)wp;
    uint4*       dst = (uint4*)smem;
    for (int i = tid; i < SMEM_BYTES / 16; i += TPB) dst[i] = src[i];
  }
  __syncthreads();

  const long tok0 = (long)blockIdx.x * TOK_PER_WG + (long)wave * (GBLK * 16);

  // A fragments (16-bit A 16x32): lane -> M = lane&15, halves:
  //   K = (lane>=16 ? 16:0) + h   -> one contiguous
  //   16-float chunk per fragment, converted f32 -> f16 here.
  v16h  A[GBLK][2];
  float bestv[GBLK][8];
  int   bestt[GBLK][8];

#pragma unroll
  for (int g = 0; g < GBLK; ++g) {
    const float* xr = x + (tok0 + g * 16 + lan16) * EMBED;
#pragma unroll
    for (int kk = 0; kk < 2; ++kk) {
      const f32x4 c0 = *(const f32x4*)(xr + kk * 32 + koff);
      const f32x4 c1 = *(const f32x4*)(xr + kk * 32 + koff + 4);
      const f32x4 c2 = *(const f32x4*)(xr + kk * 32 + koff + 8);
      const f32x4 c3 = *(const f32x4*)(xr + kk * 32 + koff + 12);
      v16h a;
      a[0]  = (_Float16)c0.x; a[1]  = (_Float16)c0.y;
      a[2]  = (_Float16)c0.z; a[3]  = (_Float16)c0.w;
      a[4]  = (_Float16)c1.x; a[5]  = (_Float16)c1.y;
      a[6]  = (_Float16)c1.z; a[7]  = (_Float16)c1.w;
      a[8]  = (_Float16)c2.x; a[9]  = (_Float16)c2.y;
      a[10] = (_Float16)c2.z; a[11] = (_Float16)c2.w;
      a[12] = (_Float16)c3.x; a[13] = (_Float16)c3.y;
      a[14] = (_Float16)c3.z; a[15] = (_Float16)c3.w;
      A[g][kk] = a;
    }
#pragma unroll
    for (int e = 0; e < 8; ++e) { bestv[g][e] = -3.0e38f; bestt[g][e] = 0; }
  }

  // Sweep all 64 code tiles; 2 WMMAs (K=64) per 16x16 score tile.
  // Score tile element e of lane: row M = e + 8*(lane/16), col n = lane&15.
  for (int t = 0; t < NTILES; ++t) {
    const v16h B0   = *(const v16h*)(lb + ((t * 2 + 0) * 32 + lane) * 16);
    const v16h B1   = *(const v16h*)(lb + ((t * 2 + 1) * 32 + lane) * 16);
    const float bias = lsq[t * 16 + lan16];
#pragma unroll
    for (int g = 0; g < GBLK; ++g) {
      v8f acc = {};
      acc = __builtin_amdgcn_wmma_f32_16x16x32_f16(false, A[g][0], false, B0,
                                                   (short)0, acc, false, false);
      acc = __builtin_amdgcn_wmma_f32_16x16x32_f16(false, A[g][1], false, B1,
                                                   (short)0, acc, false, false);
#pragma unroll
      for (int e = 0; e < 8; ++e) {
        const float s = acc[e] - bias;   // score = x.w_j - 0.5*||w_j||^2
        if (s > bestv[g][e]) { bestv[g][e] = s; bestt[g][e] = t; }
      }
    }
  }

  // Per-row argmax across the 16 lanes of each half-wave, then gather the
  // winning codebook column (exact fp32 w, L2-resident) to the output.
#pragma unroll
  for (int g = 0; g < GBLK; ++g) {
    int kwin[8];
#pragma unroll
    for (int e = 0; e < 8; ++e) {
      float bv = bestv[g][e];
      int   bi = bestt[g][e] * 16 + lan16;   // global code index
#pragma unroll
      for (int m = 1; m < 16; m <<= 1) {     // butterfly within each half
        const float ov = __shfl_xor(bv, m, 32);
        const int   oi = __shfl_xor(bi, m, 32);
        if (ov > bv || (ov == bv && oi < bi)) { bv = ov; bi = oi; }
      }
      kwin[e] = bi;                          // winner for row e + 8*(lane/16)
    }
    const long orow0 = (tok0 + g * 16) * EMBED;
#pragma unroll
    for (int m2 = 0; m2 < 16; ++m2) {
      const int k = __shfl(kwin[m2 & 7], (m2 < 8) ? 0 : 16, 32);
      out[orow0 + m2 * EMBED + lane]      = w[lane * NCODE + k];
      out[orow0 + m2 * EMBED + lane + 32] = w[(lane + 32) * NCODE + k];
    }
  }
}

// ---------------------------------------------------------------------------
extern "C" void kernel_launch(void* const* d_in, const int* in_sizes, int n_in,
                              void* d_out, int out_size, void* d_ws,
                              size_t ws_size, hipStream_t stream) {
  (void)in_sizes; (void)n_in; (void)out_size; (void)ws_size;
  const float* x = (const float*)d_in[0];   // [262144, 64] f32
  const float* w = (const float*)d_in[1];   // [64, 1024]  f32
  float*     out = (float*)d_out;           // [262144, 64] f32
  _Float16*   wp = (_Float16*)d_ws;                       // 128 KiB packed B
  float*     wsq = (float*)((char*)d_ws + WP_BYTES);      // 4 KiB biases

  (void)hipFuncSetAttribute((const void*)vq_main_kernel,
                            hipFuncAttributeMaxDynamicSharedMemorySize,
                            SMEM_BYTES);

  vq_pack_kernel<<<WP_HALFS / TPB, TPB, 0, stream>>>(w, wp, wsq);
  vq_main_kernel<<<NTOK / TOK_PER_WG, TPB, SMEM_BYTES, stream>>>(x, w, wp, out);
}
